// TabPDLHead_45612552684160
// MI455X (gfx1250) — matrix operand order repaired and checked by reading, not compiled, not run on hardware
//
#include <hip/hip_runtime.h>
#include <hip/hip_bf16.h>
#include <math.h>

// ---------------------------------------------------------------------------
// TabPDLHead fused kernel set for gfx1250 (MI455X, wave32, WMMA)
//   B=8, N=4096, M=2048, D=512, C=10
// ---------------------------------------------------------------------------

typedef __bf16 bf16_t;
typedef __attribute__((ext_vector_type(16))) __bf16 v16bf;
typedef __attribute__((ext_vector_type(8)))  float  v8f;

#define B_   8
#define N_   4096
#define M_   2048
#define D_   512
#define C_   10
#define LDSPAD 520   // row stride (bf16 elems) for LDS tiles; 520*2 = 1040 = 65*16 bytes

// round-to-nearest-even f32 -> bf16
static __device__ __forceinline__ bf16_t f2bf(float f) {
    unsigned u = __builtin_bit_cast(unsigned, f);
    unsigned r = u + 0x7FFFu + ((u >> 16) & 1u);
    unsigned short h = (unsigned short)(r >> 16);
    return __builtin_bit_cast(bf16_t, h);
}

// 16 contiguous bf16 (32 bytes)
static __device__ __forceinline__ v16bf load_v16bf(const bf16_t* p) {
    union { v16bf v; uint4 q[2]; } u;
    const uint4* s = (const uint4*)p;
    u.q[0] = s[0];
    u.q[1] = s[1];
    return u.v;
}

// two disjoint 8-element chunks (A-fragment use here: K 16h..16h+7 / 16h+8..16h+15 for lane half h)
static __device__ __forceinline__ v16bf load_v16bf_2x(const bf16_t* lo, const bf16_t* hi) {
    union { v16bf v; uint4 q[2]; } u;
    u.q[0] = *(const uint4*)lo;
    u.q[1] = *(const uint4*)hi;
    return u.v;
}

static __device__ __forceinline__ v8f wmma_bf16(v16bf a, v16bf b, v8f c) {
    return __builtin_amdgcn_wmma_f32_16x16x32_bf16(
        /*neg_a=*/false, a, /*neg_b=*/false, b,
        /*c_mod=*/(short)0, c, /*reuse_a=*/false, /*reuse_b=*/false);
}

// CDNA5 async memory->LDS copy (ASYNCcnt path, §15.18.3 op 98)
static __device__ __forceinline__ void async_load_b128(unsigned lds_off, const void* gptr) {
    asm volatile("global_load_async_to_lds_b128 %0, %1, off"
                 :: "v"(lds_off), "v"(gptr) : "memory");
}
static __device__ __forceinline__ void wait_asynccnt0() {
    asm volatile("s_wait_asynccnt 0x0" ::: "memory");
}

// ---------------------------------------------------------------------------
// Kernel 0: convert Wq / Wk (512x512 f32, row-major (e,d)) to bf16.
// Keeping (e,d) row-major means a WMMA B-fragment column (fixed e, contiguous d)
// is a single contiguous 32-byte per-lane load.
// ---------------------------------------------------------------------------
__global__ void wconv_kernel(const float* __restrict__ Wq, const float* __restrict__ Wk,
                             bf16_t* __restrict__ Wqb, bf16_t* __restrict__ Wkb) {
    int i = blockIdx.x * 256 + threadIdx.x;
    if (i < D_ * D_) {
        Wqb[i] = f2bf(Wq[i]);
        Wkb[i] = f2bf(Wk[i]);
    }
}

// ---------------------------------------------------------------------------
// Kernel 1/2: fused LayerNorm + projection  O = LN(H) @ W^T   (bf16 out)
//   block = 256 threads (8 waves): 32 rows x 256 cols of output per block,
//   gridDim.y = 2 selects the column half.
//   Phase 1: each wave LayerNorms 4 rows into LDS (bf16).
//   Phase 2: wave (rg,ct) computes a 16x64 tile: 4 independent accumulator
//            chains, software-pipelined A/B fragment loads (prefetch next
//            K-chunk while current chunk's 4 WMMAs issue).
// ---------------------------------------------------------------------------
__global__ __launch_bounds__(256)
void ln_proj_kernel(const float* __restrict__ H,
                    const float* __restrict__ lnw, const float* __restrict__ lnb,
                    const bf16_t* __restrict__ Wb,
                    bf16_t* __restrict__ Ob, int R) {
    __shared__ bf16_t At[32 * LDSPAD];

    const int tid  = threadIdx.x;
    const int lane = tid & 31;
    const int w    = tid >> 5;
    const int rowbase = blockIdx.x * 32;

    // ----- Phase 1: LayerNorm 4 rows per wave -----
    for (int rr = 0; rr < 4; ++rr) {
        const int r    = 4 * w + rr;
        const int grow = rowbase + r;
        if (grow < R) {
            const float* src = H + (size_t)grow * D_ + lane * 16;
            float xr[16];
            #pragma unroll
            for (int q4 = 0; q4 < 4; ++q4) {
                float4 x = ((const float4*)src)[q4];
                xr[q4 * 4 + 0] = x.x; xr[q4 * 4 + 1] = x.y;
                xr[q4 * 4 + 2] = x.z; xr[q4 * 4 + 3] = x.w;
            }
            float s = 0.f, sq = 0.f;
            #pragma unroll
            for (int i = 0; i < 16; ++i) { s += xr[i]; sq += xr[i] * xr[i]; }
            #pragma unroll
            for (int o = 16; o > 0; o >>= 1) {
                s  += __shfl_xor(s,  o, 32);
                sq += __shfl_xor(sq, o, 32);
            }
            const float mean = s * (1.0f / D_);
            const float var  = sq * (1.0f / D_) - mean * mean;
            const float rstd = rsqrtf(var + 1e-5f);

            const float* wp = lnw + lane * 16;
            const float* bp = lnb + lane * 16;
            #pragma unroll
            for (int q4 = 0; q4 < 4; ++q4) {
                float4 wv = ((const float4*)wp)[q4];
                float4 bv = ((const float4*)bp)[q4];
                const int d = lane * 16 + q4 * 4;
                At[r * LDSPAD + d + 0] = f2bf((xr[q4*4+0] - mean) * rstd * wv.x + bv.x);
                At[r * LDSPAD + d + 1] = f2bf((xr[q4*4+1] - mean) * rstd * wv.y + bv.y);
                At[r * LDSPAD + d + 2] = f2bf((xr[q4*4+2] - mean) * rstd * wv.z + bv.z);
                At[r * LDSPAD + d + 3] = f2bf((xr[q4*4+3] - mean) * rstd * wv.w + bv.w);
            }
        }
    }
    __syncthreads();

    // ----- Phase 2: WMMA GEMM, 16x64 per wave, software pipelined -----
    const int rg = w >> 2;                       // row-tile group (0..1)
    const int ct = w & 3;                        // col-tile group (0..3)
    const int ml = lane & 15;                    // fragment row/col within tile
    const int hf = lane >> 4;                    // lane half
    const int colbase = blockIdx.y * 256 + ct * 64;

    v8f acc[4];
    #pragma unroll
    for (int t = 0; t < 4; ++t)
        #pragma unroll
        for (int j = 0; j < 8; ++j) acc[t][j] = 0.0f;

    const bf16_t* arow  = &At[(rg * 16 + ml) * LDSPAD + hf * 16];
    const bf16_t* wbase = Wb + (size_t)(colbase + ml) * D_ + hf * 16;

    v16bf a  = load_v16bf_2x(arow, arow + 8);
    v16bf b0 = load_v16bf(wbase + 0 * 16 * D_);
    v16bf b1 = load_v16bf(wbase + 1 * 16 * D_);
    v16bf b2 = load_v16bf(wbase + 2 * 16 * D_);
    v16bf b3 = load_v16bf(wbase + 3 * 16 * D_);

    #pragma unroll
    for (int kc = 0; kc < 16; ++kc) {
        v16bf an, c0, c1, c2, c3;
        if (kc < 15) {                      // prefetch next K-chunk
            const int d1 = (kc + 1) * 32;
            an = load_v16bf_2x(arow + d1, arow + d1 + 8);
            c0 = load_v16bf(wbase + d1 + 0 * 16 * D_);
            c1 = load_v16bf(wbase + d1 + 1 * 16 * D_);
            c2 = load_v16bf(wbase + d1 + 2 * 16 * D_);
            c3 = load_v16bf(wbase + d1 + 3 * 16 * D_);
        }
        acc[0] = wmma_bf16(a, b0, acc[0]);
        acc[1] = wmma_bf16(a, b1, acc[1]);
        acc[2] = wmma_bf16(a, b2, acc[2]);
        acc[3] = wmma_bf16(a, b3, acc[3]);
        if (kc < 15) { a = an; b0 = c0; b1 = c1; b2 = c2; b3 = c3; }
    }

    // store 16x64 tile as bf16 (C/D layout: VGPR j -> row j + 8*half, lane%16 -> col)
    const int mrow = rowbase + rg * 16 + 8 * hf;
    #pragma unroll
    for (int t = 0; t < 4; ++t) {
        const int col = colbase + t * 16 + ml;
        #pragma unroll
        for (int j = 0; j < 8; ++j)
            Ob[(size_t)(mrow + j) * D_ + col] = f2bf(acc[t][j]);
    }
}

// ---------------------------------------------------------------------------
// Kernel 3: fused  logits = tau*Q@K^T + bias -> sigmoid -> class pool -> logP
//   grid = (M/16, B); block = 256 threads (8 waves).
//   Q tile (16 x 512 bf16) staged in LDS via global_load_async_to_lds_b128.
//   Each wave processes 4 n-tiles at a time (4 independent WMMA chains),
//   software-pipelined K fragment loads; sigmoid values scattered into a
//   16x16 LDS class-pool with ds_add_f32 atomics. Logits never touch HBM.
// ---------------------------------------------------------------------------
__global__ __launch_bounds__(256)
void fused_attn_pool_kernel(const bf16_t* __restrict__ Qb, const bf16_t* __restrict__ Kb,
                            const int* __restrict__ y, const unsigned char* __restrict__ mask,
                            const float* __restrict__ tau_p, const float* __restrict__ bias_p,
                            float* __restrict__ out) {
    __shared__ bf16_t Qt[16 * LDSPAD];
    __shared__ float  pool[16 * 16];

    const int tid  = threadIdx.x;
    const int lane = tid & 31;
    const int w    = tid >> 5;
    const int b    = blockIdx.y;
    const int mbase = blockIdx.x * 16;

    // stage Q tile (16 rows x 512 bf16 = 1024 x b128) via async copy to LDS
    const uint4* qsrc = (const uint4*)(Qb + ((size_t)(b * M_ + mbase)) * D_);
    #pragma unroll
    for (int i = 0; i < 4; ++i) {
        const int f = tid + 256 * i;         // b128 chunk id
        const int r = f >> 6;                // 64 chunks per row
        const int c = f & 63;
        const unsigned lds_off = (unsigned)(uintptr_t)&Qt[r * LDSPAD + c * 8];
        async_load_b128(lds_off, (const void*)(qsrc + f));
    }
    pool[tid] = 0.0f;
    wait_asynccnt0();
    __syncthreads();

    const float tp   = tau_p[0];
    const float tau  = (tp > 20.0f ? tp : log1pf(__expf(tp))) + 1e-6f;
    const float bias = bias_p[0];

    const int nl = lane & 15;     // fragment row (A) / col (B) index
    const int hf = lane >> 4;     // lane half
    const bf16_t* arow = &Qt[nl * LDSPAD + hf * 16];

    for (int it = 0; it < 8; ++it) {
        const int nt0   = (it * 8 + w) * 4;          // first of 4 n-tiles
        const int nbase = nt0 * 16;

        v8f acc[4];
        #pragma unroll
        for (int t = 0; t < 4; ++t)
            #pragma unroll
            for (int j = 0; j < 8; ++j) acc[t][j] = 0.0f;

        const bf16_t* kbase = Kb + ((size_t)(b * N_ + nbase + nl)) * D_ + hf * 16;

        v16bf a  = load_v16bf_2x(arow, arow + 8);
        v16bf b0 = load_v16bf(kbase + 0 * 16 * D_);
        v16bf b1 = load_v16bf(kbase + 1 * 16 * D_);
        v16bf b2 = load_v16bf(kbase + 2 * 16 * D_);
        v16bf b3 = load_v16bf(kbase + 3 * 16 * D_);

        #pragma unroll
        for (int kc = 0; kc < 16; ++kc) {
            v16bf an, c0, c1, c2, c3;
            if (kc < 15) {                  // prefetch next K-chunk
                const int d1 = (kc + 1) * 32;
                an = load_v16bf_2x(arow + d1, arow + d1 + 8);
                c0 = load_v16bf(kbase + d1 + 0 * 16 * D_);
                c1 = load_v16bf(kbase + d1 + 1 * 16 * D_);
                c2 = load_v16bf(kbase + d1 + 2 * 16 * D_);
                c3 = load_v16bf(kbase + d1 + 3 * 16 * D_);
            }
            acc[0] = wmma_bf16(a, b0, acc[0]);
            acc[1] = wmma_bf16(a, b1, acc[1]);
            acc[2] = wmma_bf16(a, b2, acc[2]);
            acc[3] = wmma_bf16(a, b3, acc[3]);
            if (kc < 15) { a = an; b0 = c0; b1 = c1; b2 = c2; b3 = c3; }
        }

        // pool: lane owns column n = nbase + t*16 + nl, rows m = j + 8*hf
        #pragma unroll
        for (int t = 0; t < 4; ++t) {
            const int   ng  = b * N_ + nbase + t * 16 + nl;
            const int   cls = y[ng];
            const float sc  = mask[ng] ? 1.0f : 0.0f;  // masked -> sigmoid(-inf)=0
            #pragma unroll
            for (int j = 0; j < 8; ++j) {
                const float logit = tau * acc[t][j] + bias;
                const float g = sc / (1.0f + __expf(-logit));
                atomicAdd(&pool[(j + 8 * hf) * 16 + cls], g);   // ds_add_f32
            }
        }
    }
    __syncthreads();

    // finalize: normalize over classes, clamp, log
    if (tid < 16 * C_) {
        const int m = tid / C_;
        const int c = tid % C_;
        float rs = 0.0f;
        #pragma unroll
        for (int cc = 0; cc < C_; ++cc) rs += pool[m * 16 + cc];
        const float P     = pool[m * 16 + c];
        const float denom = fmaxf(rs, 1e-12f);
        const float val   = __logf(fmaxf(P / denom, 1e-12f));
        out[(size_t)((b * M_ + mbase + m)) * C_ + c] = val;
    }
}

// ---------------------------------------------------------------------------
// launch
// ---------------------------------------------------------------------------
extern "C" void kernel_launch(void* const* d_in, const int* in_sizes, int n_in,
                              void* d_out, int out_size, void* d_ws, size_t ws_size,
                              hipStream_t stream) {
    const float*         H_support = (const float*)d_in[0];   // (B,N,D)
    const float*         H_query   = (const float*)d_in[1];   // (B,M,D)
    const int*           y_support = (const int*)d_in[2];     // (B,N)
    const unsigned char* smask     = (const unsigned char*)d_in[3]; // (B,N) bool
    const float*         ln_w      = (const float*)d_in[4];
    const float*         ln_b      = (const float*)d_in[5];
    const float*         Wq        = (const float*)d_in[6];   // (D,D)
    const float*         Wk        = (const float*)d_in[7];   // (D,D)
    const float*         tau_p     = (const float*)d_in[8];
    const float*         bias_p    = (const float*)d_in[9];

    char* ws = (char*)d_ws;
    // workspace layout (bytes)
    bf16_t* Kb  = (bf16_t*)(ws);                      // B*N*D bf16 = 32 MB
    bf16_t* Qb  = (bf16_t*)(ws + (size_t)33554432);   // B*M*D bf16 = 16 MB
    bf16_t* Wqb = (bf16_t*)(ws + (size_t)50331648);   // 512 KB
    bf16_t* Wkb = (bf16_t*)(ws + (size_t)50855936);   // 512 KB

    // 0) weights -> bf16
    wconv_kernel<<<(D_ * D_ + 255) / 256, 256, 0, stream>>>(Wq, Wk, Wqb, Wkb);

    // 1) Q = LN(H_query) @ Wq^T   (R = B*M = 16384 rows, 2 column halves)
    ln_proj_kernel<<<dim3((B_ * M_) / 32, 2), 256, 0, stream>>>(
        H_query, ln_w, ln_b, Wqb, Qb, B_ * M_);

    // 2) K = LN(H_support) @ Wk^T (R = B*N = 32768 rows, 2 column halves)
    ln_proj_kernel<<<dim3((B_ * N_) / 32, 2), 256, 0, stream>>>(
        H_support, ln_w, ln_b, Wkb, Kb, B_ * N_);

    // 3) fused logits/sigmoid/class-pool/log
    fused_attn_pool_kernel<<<dim3(M_ / 16, B_), 256, 0, stream>>>(
        Qb, Kb, y_support, smask, tau_p, bias_p, (float*)d_out);
}
